// BruteForceMoELinear_60679297957911
// MI455X (gfx1250) — matrix operand order repaired and checked by reading, not compiled, not run on hardware
//
#include <hip/hip_runtime.h>
#include <hip/hip_bf16.h>

// ---------------------------------------------------------------------------
// MoE FFN (E=16, D=1024, H=2048, K=2, T=2048), fp32 in/out.
// bf16 WMMA (v_wmma_f32_16x16x32_bf16), fp32 accumulation.
// Weights are pre-converted fp32->bf16 ONCE per launch into d_ws (384MB of
// HBM traffic ~16.5us @ 23.3TB/s), so the GEMM hot loop is pure load+WMMA
// with no conversion VALU and only a 128MB bf16 weight stream (L2-assisted
// across the ~16 row-tile blocks per expert). Falls back to inline
// conversion if ws_size is too small.
// ---------------------------------------------------------------------------

#define E_   16
#define D_   1024
#define H_   2048
#define K_   2
#define T_   2048
#define CAP  (T_ * K_)         // worst-case rows per expert
#define ROWS 16                // token rows per workgroup (one WMMA M tile)
#define MAXTILES (CAP / ROWS)  // 256 tiles per expert worst case
#define W1ELEMS ((long long)E_ * H_ * D_)   // 32M
#define W2ELEMS ((long long)E_ * D_ * H_)   // 32M

typedef __bf16 bf16_t;
typedef bf16_t v16bf __attribute__((ext_vector_type(16)));
typedef bf16_t bf16x2 __attribute__((ext_vector_type(2)));
typedef float  v8f   __attribute__((ext_vector_type(8)));
typedef unsigned int u32x4 __attribute__((ext_vector_type(4)));

// Pack two fp32 into one dword of two bf16 (lo=a, hi=b), RNE.
__device__ __forceinline__ unsigned int pk_bf16(float a, float b) {
#if __has_builtin(__builtin_amdgcn_cvt_pk_bf16_f32)
  bf16x2 p = __builtin_amdgcn_cvt_pk_bf16_f32(a, b);
  return __builtin_bit_cast(unsigned int, p);
#else
  bf16x2 p;
  p.x = (bf16_t)a;   // fptrunc, RNE; backend may select v_cvt_pk_bf16_f32
  p.y = (bf16_t)b;
  return __builtin_bit_cast(unsigned int, p);
#endif
}

__device__ __forceinline__ unsigned short f2bf(float f) {
  bf16_t h = (bf16_t)f;
  return __builtin_bit_cast(unsigned short, h);
}

// Branch-free tanh-approx gelu (matches jax.nn.gelu approximate=True).
// tanh(y) = 1 - 2/(1+exp(2y)): native v_exp_f32 + v_rcp_f32, no libm branches.
__device__ __forceinline__ float gelu_tanh(float x) {
  const float c = 0.7978845608028654f;  // sqrt(2/pi)
  float y = c * (x + 0.044715f * x * x * x);
  float e = __expf(2.0f * y);
  float r = __builtin_amdgcn_rcpf(1.0f + e);   // e=inf -> r=0 -> tanh=1
  float th = 1.0f - 2.0f * r;
  return 0.5f * x * (1.0f + th);
}

// A fragment (16 x K=32 bf16) from LDS, row-major [16][ld] ushort.
// Lane L -> row M=L&15; kb=(L<16)?0:16; elems 0..15 = K kb..kb+15
// contiguous.  Two aligned 16B LDS loads.
__device__ __forceinline__ v16bf load_a_lds(const unsigned short* base, int ld,
                                            int k0, int lane) {
  int m  = lane & 15;
  int kb = (lane < 16) ? 0 : 16;
  const unsigned short* p = base + m * ld + k0 + kb;
  union { u32x4 q[2]; v16bf v; } u;
  u.q[0] = *(const u32x4*)(p);
  u.q[1] = *(const u32x4*)(p + 8);
  return u.v;
}

// B fragment (K=32 x 16) from PRE-CONVERTED bf16 weights, row-major [N][ld].
// Lane L -> col N=L&15; kb=(L<16)?0:16; elems 0..15 = K kb..kb+15 contiguous.
// Two aligned global_load_b128, zero conversion VALU.
__device__ __forceinline__ v16bf load_b_bf16(const unsigned short* wbase, int ld,
                                             int n0, int k0, int lane) {
  int n  = n0 + (lane & 15);
  int kb = (lane < 16) ? 0 : 16;
  const unsigned short* p = wbase + (long long)n * ld + k0 + kb;
  __builtin_prefetch(p + 128, 0, 1);           // global_prefetch_b8
  union { u32x4 q[2]; v16bf v; } u;
  u.q[0] = *(const u32x4*)(p);
  u.q[1] = *(const u32x4*)(p + 8);
  return u.v;
}

// B fragment from fp32 weights with inline conversion (fallback path).
__device__ __forceinline__ v16bf load_b_f32(const float* wbase, int ld,
                                            int n0, int k0, int lane) {
  int n  = n0 + (lane & 15);
  int kb = (lane < 16) ? 0 : 16;
  const float* p = wbase + (long long)n * ld + k0 + kb;
  __builtin_prefetch(p + 64, 0, 1);
  union { unsigned int w[8]; v16bf v; } u;
#pragma unroll
  for (int i = 0; i < 8; ++i) u.w[i] = pk_bf16(p[2 * i], p[2 * i + 1]);
  return u.v;
}

// ---------------------------------------------------------------------------
__global__ __launch_bounds__(256) void moe_init(int* counts, float* out) {
  int i = blockIdx.x * blockDim.x + threadIdx.x;
  if (i < E_) counts[i] = 0;
  const long long n = (long long)T_ * D_;
  for (long long j = i; j < n; j += (long long)gridDim.x * blockDim.x)
    out[j] = 0.0f;
}

// fp32 -> bf16 weight pre-conversion, 8 elems/thread/iter, b128 in / b128 out.
__global__ __launch_bounds__(256) void moe_cvt(const float* __restrict__ src,
                                               unsigned short* __restrict__ dst,
                                               long long n) {
  long long stride = (long long)gridDim.x * blockDim.x * 8;
  for (long long i = ((long long)blockIdx.x * blockDim.x + threadIdx.x) * 8;
       i < n; i += stride) {
    float4 f0 = *(const float4*)(src + i);
    float4 f1 = *(const float4*)(src + i + 4);
    u32x4 w;
    w.x = pk_bf16(f0.x, f0.y);
    w.y = pk_bf16(f0.z, f0.w);
    w.z = pk_bf16(f1.x, f1.y);
    w.w = pk_bf16(f1.z, f1.w);
    *(u32x4*)(dst + i) = w;
  }
}

__global__ __launch_bounds__(256) void moe_gather(const int* __restrict__ gidx,
                                                  int* counts, int* lists) {
  int i = blockIdx.x * blockDim.x + threadIdx.x;
  if (i < T_ * K_) {
    int e = gidx[i];                       // 0..15
    int pos = atomicAdd(&counts[e], 1);
    lists[e * CAP + pos] = i;              // i = token*K + k
  }
}

// One block = (expert e, 16-row tile). 8 waves. Full D output, full H hidden.
template <bool PRE>
__global__ __launch_bounds__(256) void moe_ffn(
    const float* __restrict__ inp, const float* __restrict__ gate_score,
    const float* __restrict__ w1f, const float* __restrict__ b1,
    const float* __restrict__ w2f, const float* __restrict__ b2,
    const unsigned short* __restrict__ w1b, const unsigned short* __restrict__ w2b,
    const int* __restrict__ counts, const int* __restrict__ lists,
    float* __restrict__ out) {
  __shared__ unsigned short x_lds[ROWS * D_];   // 32 KB bf16 activations
  __shared__ unsigned short h_lds[ROWS * H_];   // 64 KB bf16 hidden
  __shared__ int   row_token[ROWS];
  __shared__ float row_scale[ROWS];

  const int e    = blockIdx.x >> 8;             // MAXTILES == 256
  const int tile = blockIdx.x & (MAXTILES - 1);
  const int cnt  = counts[e];
  if (tile * ROWS >= cnt) return;               // uniform exit, EXEC stays full

  const int tid   = threadIdx.x;
  const int lane  = tid & 31;
  const int wave  = tid >> 5;                   // 0..7
  const int ln    = lane & 15;                  // N index within C/D tile
  const int mbase = (lane < 16) ? 0 : 8;        // C/D: vgpr i -> M = mbase+i

  if (tid < ROWS) {
    int idx = tile * ROWS + tid;
    if (idx < cnt) {
      int pair = lists[e * CAP + idx];
      row_token[tid] = pair >> 1;               // token
      row_scale[tid] = gate_score[pair];        // gate_score[token][k]
    } else {
      row_token[tid] = -1;
      row_scale[tid] = 0.0f;
    }
  }
  __syncthreads();

  // Gather 16 token rows -> LDS bf16 (packed b32 stores). 16 thr/row, 64 ea.
  {
    int r  = tid >> 4;
    int c0 = (tid & 15) * 64;
    int tok = row_token[r];
    const float* src = inp + (long long)(tok < 0 ? 0 : tok) * D_ + c0;
    unsigned int* dst = (unsigned int*)(x_lds + r * D_ + c0);
#pragma unroll 4
    for (int i = 0; i < 64; i += 4) {
      float4 f = *(const float4*)(src + i);
      dst[i / 2]     = pk_bf16(f.x, f.y);
      dst[i / 2 + 1] = pk_bf16(f.z, f.w);
    }
  }
  __syncthreads();

  // -------- Stage A: h = gelu(X @ w1[e]^T + b1[e]),  [16 x 2048] ----------
  // 2 N-tiles per iteration share one A fragment -> 2 independent WMMA chains.
  const float*          w1fe = w1f + (long long)e * H_ * D_;
  const unsigned short* w1be = w1b + (long long)e * H_ * D_;
  const float*          b1e  = b1 + (long long)e * H_;
  for (int j = 0; j < 8; ++j) {                 // 16 N-tiles per wave, in pairs
    const int h0 = (wave + 8 * (2 * j)) * 16;
    const int h1 = (wave + 8 * (2 * j + 1)) * 16;
    v8f acc0 = {}, acc1 = {};
    for (int k0 = 0; k0 < D_; k0 += 32) {
      v16bf a = load_a_lds(x_lds, D_, k0, lane);
      v16bf b0, b1v;
      if constexpr (PRE) {
        b0  = load_b_bf16(w1be, D_, h0, k0, lane);
        b1v = load_b_bf16(w1be, D_, h1, k0, lane);
      } else {
        b0  = load_b_f32(w1fe, D_, h0, k0, lane);
        b1v = load_b_f32(w1fe, D_, h1, k0, lane);
      }
      acc0 = __builtin_amdgcn_wmma_f32_16x16x32_bf16(
          false, a, false, b0, (short)0, acc0, false, false);
      acc1 = __builtin_amdgcn_wmma_f32_16x16x32_bf16(
          false, a, false, b1v, (short)0, acc1, false, false);
    }
    const float bias0 = b1e[h0 + ln];
    const float bias1 = b1e[h1 + ln];
#pragma unroll
    for (int i = 0; i < 8; ++i) {
      h_lds[(mbase + i) * H_ + h0 + ln] = f2bf(gelu_tanh(acc0[i] + bias0));
      h_lds[(mbase + i) * H_ + h1 + ln] = f2bf(gelu_tanh(acc1[i] + bias1));
    }
  }
  __syncthreads();

  // -------- Stage B: y = h @ w2[e]^T + b2[e], scaled atomic scatter -------
  const float*          w2fe = w2f + (long long)e * D_ * H_;
  const unsigned short* w2be = w2b + (long long)e * D_ * H_;
  const float*          b2e  = b2 + (long long)e * D_;
  for (int j = 0; j < 4; ++j) {                 // 8 N-tiles per wave, in pairs
    const int d0 = (wave + 8 * (2 * j)) * 16;
    const int d1 = (wave + 8 * (2 * j + 1)) * 16;
    v8f acc0 = {}, acc1 = {};
    for (int k0 = 0; k0 < H_; k0 += 32) {
      v16bf a = load_a_lds(h_lds, H_, k0, lane);
      v16bf b0, b1v;
      if constexpr (PRE) {
        b0  = load_b_bf16(w2be, H_, d0, k0, lane);
        b1v = load_b_bf16(w2be, H_, d1, k0, lane);
      } else {
        b0  = load_b_f32(w2fe, H_, d0, k0, lane);
        b1v = load_b_f32(w2fe, H_, d1, k0, lane);
      }
      acc0 = __builtin_amdgcn_wmma_f32_16x16x32_bf16(
          false, a, false, b0, (short)0, acc0, false, false);
      acc1 = __builtin_amdgcn_wmma_f32_16x16x32_bf16(
          false, a, false, b1v, (short)0, acc1, false, false);
    }
    const float bias0 = b2e[d0 + ln];
    const float bias1 = b2e[d1 + ln];
#pragma unroll
    for (int i = 0; i < 8; ++i) {
      int m  = mbase + i;
      int tok = row_token[m];
      if (tok >= 0) {
        float s = row_scale[m];
        unsafeAtomicAdd(&out[(long long)tok * D_ + d0 + ln], s * (acc0[i] + bias0));
        unsafeAtomicAdd(&out[(long long)tok * D_ + d1 + ln], s * (acc1[i] + bias1));
      }
    }
  }
}

// ---------------------------------------------------------------------------
extern "C" void kernel_launch(void* const* d_in, const int* in_sizes, int n_in,
                              void* d_out, int out_size, void* d_ws,
                              size_t ws_size, hipStream_t stream) {
  const float* inp        = (const float*)d_in[0];
  const int*   gate_idx   = (const int*)d_in[1];
  const float* gate_score = (const float*)d_in[2];
  const float* w1         = (const float*)d_in[3];
  const float* b1         = (const float*)d_in[4];
  const float* w2         = (const float*)d_in[5];
  const float* b2         = (const float*)d_in[6];
  float* out = (float*)d_out;

  char* base = (char*)d_ws;
  int* counts = (int*)base;                       // 16 ints (256B pad)
  int* lists  = (int*)(base + 256);               // E*CAP ints = 256 KB
  const size_t fixed = 256 + (size_t)E_ * CAP * 4;
  unsigned short* w1b = (unsigned short*)(base + fixed);
  unsigned short* w2b = w1b + W1ELEMS;
  const size_t need = fixed + (size_t)(W1ELEMS + W2ELEMS) * 2;
  const bool pre = (ws_size >= need);

  moe_init<<<256, 256, 0, stream>>>(counts, out);
  if (pre) {
    moe_cvt<<<8192, 256, 0, stream>>>(w1, w1b, W1ELEMS);
    moe_cvt<<<8192, 256, 0, stream>>>(w2, w2b, W2ELEMS);
  }
  moe_gather<<<(T_ * K_ + 255) / 256, 256, 0, stream>>>(gate_idx, counts, lists);
  if (pre) {
    moe_ffn<true><<<E_ * MAXTILES, 256, 0, stream>>>(
        inp, gate_score, w1, b1, w2, b2, w1b, w2b, counts, lists, out);
  } else {
    moe_ffn<false><<<E_ * MAXTILES, 256, 0, stream>>>(
        inp, gate_score, w1, b1, w2, b2, w1b, w2b, counts, lists, out);
  }
  (void)in_sizes; (void)n_in; (void)out_size;
}
